// SMPL_Parameter_Regressor_20117626814851
// MI455X (gfx1250) — matrix operand order repaired and checked by reading, not compiled, not run on hardware
//
#include <hip/hip_runtime.h>
#include <hip/hip_bf16.h>

// ---------------------------------------------------------------------------
// SMPL parameter regressor on MI455X (gfx1250, wave32, WMMA).
//  - 6x GEMM: bf16 WMMA (v_wmma_f32_16x16x32_bf16), f32 accumulate
//  - 256x128 block tile, 8 waves, 64x64 per wave (16 WMMA / K-step / wave)
//  - weights pre-packed to bf16 [N][Kp] K-major (zero padded) -> guard-free,
//    async global->LDS tile fills (ASYNCcnt path)
//  - activations produced in bf16 by the BN epilogue for layers 1..5
//  - batch-axis BN via deterministic two-stage column reduction
//  - 3x3 SVD orthogonalization via Newton polar iteration
// ---------------------------------------------------------------------------

typedef __attribute__((ext_vector_type(16))) __bf16 bf16x16;
typedef __attribute__((ext_vector_type(8)))  __bf16 bf16x8;
typedef __attribute__((ext_vector_type(4)))  __bf16 bf16x4;
typedef __attribute__((ext_vector_type(8)))  float  f32x8;

#define BM 256
#define BN 128
#define BK 32
#define LDS_STRIDE 40   // padded bf16 stride: 80B rows (16B aligned), 60KB total
#define THREADS 256     // 8 wave32s, 4(M) x 2(N), 64x64 C-tile per wave

static __device__ __forceinline__ void wait_async0() {
#if __has_builtin(__builtin_amdgcn_s_wait_asynccnt)
  __builtin_amdgcn_s_wait_asynccnt(0);
#else
  asm volatile("s_wait_asynccnt 0x0" ::: "memory");
#endif
}

// One 16B chunk: global (64-bit addr) -> LDS (32-bit wave-relative addr).
static __device__ __forceinline__ void async_copy16(const void* gptr, void* lptr) {
  unsigned int la = (unsigned int)(uintptr_t)lptr;      // addr[31:0] = LDS offset
  unsigned long long ga = (unsigned long long)(uintptr_t)gptr;
  asm volatile("global_load_async_to_lds_b128 %0, %1, off"
               :: "v"(la), "v"(ga) : "memory");
}

// ---------------------------------------------------------------------------
// GEMM: C(MxN) = A @ B + bias.
//   A_BF16=true : A is bf16 [M][Kp]   (activations, pre-converted)
//   A_BF16=false: A is fp32 [M][K]    (raw input x), converted during fill
//   B is bf16 [Npad][Kp], K-major, zero padded -> no guards in any fill.
// Grid: (ceil(N/128), M/256); M multiple of 256.
// ---------------------------------------------------------------------------
template <bool A_BF16>
__global__ __launch_bounds__(THREADS)
void gemm_kernel(const void* __restrict__ Av, const __bf16* __restrict__ B,
                 const float* __restrict__ bias, float* __restrict__ C,
                 int M, int N, int K, int Kp) {
  __shared__ __bf16 sA[2][BM * LDS_STRIDE];   // 2 * 20KB
  __shared__ __bf16 sB[2][BN * LDS_STRIDE];   // 2 * 10KB

  const int tid  = threadIdx.x;
  const int lane = tid & 31;
  const int wid  = tid >> 5;       // 0..7
  const int wm   = wid >> 1;       // 0..3  -> 64 rows of C
  const int wn   = wid & 1;        // 0..1  -> 64 cols of C

  const int m0 = blockIdx.y * BM;
  const int n0 = blockIdx.x * BN;

  f32x8 acc[4][4];
#pragma unroll
  for (int i = 0; i < 4; ++i)
#pragma unroll
    for (int j = 0; j < 4; ++j)
#pragma unroll
      for (int v = 0; v < 8; ++v) acc[i][j][v] = 0.0f;

  // ---- B tile: 128 rows x 64B = 512 x 16B chunks, 2 per thread, async.
  auto fill_B = [&](int buf, int k0) {
#pragma unroll
    for (int it = 0; it < 2; ++it) {
      int c   = it * THREADS + tid;
      int row = c >> 2;
      int ch  = c & 3;                      // 16B chunk within the 64B row
      const __bf16* g = B + (size_t)(n0 + row) * Kp + k0 + ch * 8;
      async_copy16(g, &sB[buf][row * LDS_STRIDE + ch * 8]);
    }
  };

  // ---- A tile: 256 rows x 64B = 1024 x 16B chunks (bf16 path).
  auto fill_A = [&](int buf, int k0) {
    if constexpr (A_BF16) {
      const __bf16* A = (const __bf16*)Av;
#pragma unroll
      for (int it = 0; it < 4; ++it) {
        int c   = it * THREADS + tid;
        int row = c >> 2;
        int ch  = c & 3;
        const __bf16* g = A + (size_t)(m0 + row) * Kp + k0 + ch * 8;
        async_copy16(g, &sA[buf][row * LDS_STRIDE + ch * 8]);
      }
    } else {
      const float* A = (const float*)Av;
      if (k0 + BK <= K) {                       // fast path: no guards
#pragma unroll
        for (int it = 0; it < 8; ++it) {
          int e   = (it * THREADS + tid) * 4;
          int row = e >> 5;
          int col = e & 31;
          float4 v = *(const float4*)(A + (size_t)(m0 + row) * K + k0 + col);
          bf16x4 b;
          b[0] = (__bf16)v.x; b[1] = (__bf16)v.y;
          b[2] = (__bf16)v.z; b[3] = (__bf16)v.w;
          *(bf16x4*)&sA[buf][row * LDS_STRIDE + col] = b;
        }
      } else {                                  // single K-tail tile
#pragma unroll
        for (int it = 0; it < 8; ++it) {
          int e   = (it * THREADS + tid) * 4;
          int row = e >> 5;
          int col = e & 31;
          const float* src = A + (size_t)(m0 + row) * K;
          bf16x4 b;
#pragma unroll
          for (int j = 0; j < 4; ++j) {
            int gk = k0 + col + j;
            b[j] = (__bf16)((gk < K) ? src[gk] : 0.0f);
          }
          *(bf16x4*)&sA[buf][row * LDS_STRIDE + col] = b;
        }
      }
    }
  };

  const int nk = Kp / BK;
  fill_A(0, 0);
  fill_B(0, 0);

  const int mrow = lane & 15;
  const int kloA = (lane < 16) ? 0 : 8;    // CDNA5 16-bit A-frag layout
  const int kloB = (lane < 16) ? 0 : 8;    // CDNA5 16-bit B-frag layout

  for (int kt = 0; kt < nk; ++kt) {
    wait_async0();           // own async LDS writes done
    __syncthreads();         // everyone's writes visible / reads drained
    const int cur = kt & 1;
    if (kt + 1 < nk) {
      fill_A(cur ^ 1, (kt + 1) * BK);
      fill_B(cur ^ 1, (kt + 1) * BK);
    }

    const __bf16* pA = sA[cur];
    const __bf16* pB = sB[cur];

    bf16x16 afrag[4];
#pragma unroll
    for (int i = 0; i < 4; ++i) {
      int r = wm * 64 + i * 16 + mrow;
      union { bf16x16 v; bf16x8 h[2]; } u;
      u.h[0] = *(const bf16x8*)&pA[r * LDS_STRIDE + kloA];
      u.h[1] = *(const bf16x8*)&pA[r * LDS_STRIDE + kloA + 16];
      afrag[i] = u.v;
    }
    bf16x16 bfrag[4];
#pragma unroll
    for (int j = 0; j < 4; ++j) {
      int cc = wn * 64 + j * 16 + mrow;
      union { bf16x16 v; bf16x8 h[2]; } u;
      u.h[0] = *(const bf16x8*)&pB[cc * LDS_STRIDE + kloB];
      u.h[1] = *(const bf16x8*)&pB[cc * LDS_STRIDE + kloB + 16];
      bfrag[j] = u.v;
    }

#pragma unroll
    for (int i = 0; i < 4; ++i)
#pragma unroll
      for (int j = 0; j < 4; ++j)
        acc[i][j] = __builtin_amdgcn_wmma_f32_16x16x32_bf16(
            false, afrag[i], false, bfrag[j], (short)0, acc[i][j], false, false);
  }

  // Epilogue: VGPR v -> M=v (lanes 0-15) / M=8+v (lanes 16-31); N = lane&15.
  const int ncol  = lane & 15;
  const int rbase = (lane >> 4) * 8;
#pragma unroll
  for (int i = 0; i < 4; ++i) {
#pragma unroll
    for (int j = 0; j < 4; ++j) {
      int gn = n0 + wn * 64 + j * 16 + ncol;
      if (gn < N) {
        float bv = bias[gn];
#pragma unroll
        for (int v = 0; v < 8; ++v) {
          int gm = m0 + wm * 64 + i * 16 + rbase + v;
          C[(size_t)gm * N + gn] = acc[i][j][v] + bv;
        }
      }
    }
  }
}

// ---------------------------------------------------------------------------
// Weight pack: fp32 [K][N] -> bf16 [Npad][Kp] K-major, zero padded.
// ---------------------------------------------------------------------------
__global__ __launch_bounds__(256)
void pack_w_kernel(const float* __restrict__ W, __bf16* __restrict__ out,
                   int K, int N, int Kp) {
  int k = blockIdx.x * 256 + threadIdx.x;
  int n = blockIdx.y;
  if (k >= Kp) return;
  float v = (k < K && n < N) ? W[(size_t)k * N + n] : 0.0f;
  out[(size_t)n * Kp + k] = (__bf16)v;
}

// ---------------------------------------------------------------------------
// Batch-norm stats: deterministic two-stage column reduction over 8192 rows.
// ---------------------------------------------------------------------------
__global__ __launch_bounds__(256)
void bn_partial_kernel(const float* __restrict__ y, float* __restrict__ pbuf,
                       int Mrows, int N, int chunks) {
  const int tx = threadIdx.x & 63;
  const int ty = threadIdx.x >> 6;       // 0..3
  const int c  = blockIdx.x * 64 + tx;
  const int rowsPer = Mrows / chunks;
  const int r0 = blockIdx.y * rowsPer;
  float s = 0.0f, s2 = 0.0f;
  for (int r = r0 + ty; r < r0 + rowsPer; r += 4) {
    float v = y[(size_t)r * N + c];
    s += v; s2 += v * v;
  }
  __shared__ float sh[2][4][64];
  sh[0][ty][tx] = s; sh[1][ty][tx] = s2;
  __syncthreads();
  if (ty == 0) {
    s  = sh[0][0][tx] + sh[0][1][tx] + sh[0][2][tx] + sh[0][3][tx];
    s2 = sh[1][0][tx] + sh[1][1][tx] + sh[1][2][tx] + sh[1][3][tx];
    pbuf[((size_t)blockIdx.y * N + c) * 2 + 0] = s;
    pbuf[((size_t)blockIdx.y * N + c) * 2 + 1] = s2;
  }
}

__global__ __launch_bounds__(256)
void bn_finalize_kernel(const float* __restrict__ pbuf, float* __restrict__ stats,
                        int N, int chunks, int Mrows) {
  int c = blockIdx.x * 256 + threadIdx.x;
  if (c >= N) return;
  float s = 0.0f, s2 = 0.0f;
  for (int k = 0; k < chunks; ++k) {
    s  += pbuf[((size_t)k * N + c) * 2 + 0];
    s2 += pbuf[((size_t)k * N + c) * 2 + 1];
  }
  float m   = s / (float)Mrows;
  float var = s2 / (float)Mrows - m * m;
  stats[2 * c + 0] = m;
  stats[2 * c + 1] = rsqrtf(fmaxf(var, 0.0f) + 1e-5f);
}

// out = [relu]( [res +] g*(y-mean)*invstd + be ); also emits bf16 copy for
// the next GEMM's A operand.
template <bool RELU, bool RES>
__global__ __launch_bounds__(256)
void bn_apply_kernel(const float* __restrict__ y, const float* __restrict__ stats,
                     const float* __restrict__ g, const float* __restrict__ be,
                     const float* __restrict__ res, float* __restrict__ out,
                     __bf16* __restrict__ out_bf, int total, int nmask) {
  int i = blockIdx.x * 256 + threadIdx.x;
  if (i >= total) return;
  int c = i & nmask;
  float v = (y[i] - stats[2 * c]) * stats[2 * c + 1] * g[c] + be[c];
  if (RES) v += res[i];
  if (RELU) v = fmaxf(v, 0.0f);
  out[i]    = v;
  out_bf[i] = (__bf16)v;
}

// ---------------------------------------------------------------------------
// Betas: out[:, 216:226] slice copy.
// ---------------------------------------------------------------------------
__global__ __launch_bounds__(256)
void betas_kernel(const float* __restrict__ yout, float* __restrict__ outB,
                  int rows, int ldy) {
  int i = blockIdx.x * 256 + threadIdx.x;
  if (i >= rows * 10) return;
  int r = i / 10, j = i - r * 10;
  outB[i] = yout[(size_t)r * ldy + 216 + j];
}

// ---------------------------------------------------------------------------
// Polar decomposition of 3x3: R = U*Vh via X <- 0.5*(X + cof(X)/det(X));
// output R*sign(det(R)) (matches the reference's det fixup).
// ---------------------------------------------------------------------------
__global__ __launch_bounds__(256)
void polar_kernel(const float* __restrict__ yout, float* __restrict__ outR,
                  int count, int ldy) {
  int gidx = blockIdx.x * 256 + threadIdx.x;
  if (gidx >= count) return;
  int r = gidx / 24, q = gidx - r * 24;
  const float* src = yout + (size_t)r * ldy + q * 9;
  float x[9];
#pragma unroll
  for (int e = 0; e < 9; ++e) x[e] = src[e];

  float fn = 0.0f;
#pragma unroll
  for (int e = 0; e < 9; ++e) fn += x[e] * x[e];
  float sc = rsqrtf(fn + 1e-20f);
#pragma unroll
  for (int e = 0; e < 9; ++e) x[e] *= sc;

  for (int it = 0; it < 15; ++it) {
    float c00 = x[4]*x[8] - x[5]*x[7];
    float c01 = x[5]*x[6] - x[3]*x[8];
    float c02 = x[3]*x[7] - x[4]*x[6];
    float c10 = x[2]*x[7] - x[1]*x[8];
    float c11 = x[0]*x[8] - x[2]*x[6];
    float c12 = x[1]*x[6] - x[0]*x[7];
    float c20 = x[1]*x[5] - x[2]*x[4];
    float c21 = x[2]*x[3] - x[0]*x[5];
    float c22 = x[0]*x[4] - x[1]*x[3];
    float det = x[0]*c00 + x[1]*c01 + x[2]*c02;
    float inv = 1.0f / (det + copysignf(1e-12f, det));
    x[0] = 0.5f*(x[0] + c00*inv); x[1] = 0.5f*(x[1] + c01*inv); x[2] = 0.5f*(x[2] + c02*inv);
    x[3] = 0.5f*(x[3] + c10*inv); x[4] = 0.5f*(x[4] + c11*inv); x[5] = 0.5f*(x[5] + c12*inv);
    x[6] = 0.5f*(x[6] + c20*inv); x[7] = 0.5f*(x[7] + c21*inv); x[8] = 0.5f*(x[8] + c22*inv);
  }

  float d = x[0]*(x[4]*x[8]-x[5]*x[7]) - x[1]*(x[3]*x[8]-x[5]*x[6]) + x[2]*(x[3]*x[7]-x[4]*x[6]);
  float s = copysignf(1.0f, d);
  float* dst = outR + (size_t)gidx * 9;
#pragma unroll
  for (int e = 0; e < 9; ++e) dst[e] = x[e] * s;
}

// ---------------------------------------------------------------------------
extern "C" void kernel_launch(void* const* d_in, const int* in_sizes, int n_in,
                              void* d_out, int out_size, void* d_ws, size_t ws_size,
                              hipStream_t stream) {
  (void)in_sizes; (void)n_in; (void)out_size; (void)ws_size;
  const int Mrows = 8192, H = 1024, K0 = 5169, Kp0 = 5184, NOUT = 226,
            NOUTP = 256, CHUNKS = 8;

  const float* x    = (const float*)d_in[0];
  const float* W0   = (const float*)d_in[1];
  const float* b0   = (const float*)d_in[2];
  const float* g0   = (const float*)d_in[3];
  const float* be0  = (const float*)d_in[4];
  const float* W1a  = (const float*)d_in[5];
  const float* b1a  = (const float*)d_in[6];
  const float* g1a  = (const float*)d_in[7];
  const float* be1a = (const float*)d_in[8];
  const float* W1b  = (const float*)d_in[9];
  const float* b1b  = (const float*)d_in[10];
  const float* g1b  = (const float*)d_in[11];
  const float* be1b = (const float*)d_in[12];
  const float* W2a  = (const float*)d_in[13];
  const float* b2a  = (const float*)d_in[14];
  const float* g2a  = (const float*)d_in[15];
  const float* be2a = (const float*)d_in[16];
  const float* W2b  = (const float*)d_in[17];
  const float* b2b  = (const float*)d_in[18];
  const float* g2b  = (const float*)d_in[19];
  const float* be2b = (const float*)d_in[20];
  const float* W3   = (const float*)d_in[21];
  const float* b3   = (const float*)d_in[22];

  // ---- Workspace carve-out (~128 MB) ----
  char* p = (char*)d_ws;
  auto take = [&](size_t bytes) {
    char* r = p;
    p += (bytes + 255) & ~(size_t)255;
    return (void*)r;
  };
  float*  y     = (float*)take((size_t)Mrows * H * 4);
  float*  h     = (float*)take((size_t)Mrows * H * 4);
  float*  yout  = (float*)take((size_t)Mrows * NOUT * 4);
  float*  pbuf  = (float*)take((size_t)CHUNKS * H * 2 * 4);
  float*  stats = (float*)take((size_t)H * 2 * 4);
  __bf16* hbf   = (__bf16*)take((size_t)Mrows * H * 2);
  __bf16* tbf   = (__bf16*)take((size_t)Mrows * H * 2);
  __bf16* w0p   = (__bf16*)take((size_t)H * Kp0 * 2);
  __bf16* w1ap  = (__bf16*)take((size_t)H * H * 2);
  __bf16* w1bp  = (__bf16*)take((size_t)H * H * 2);
  __bf16* w2ap  = (__bf16*)take((size_t)H * H * 2);
  __bf16* w2bp  = (__bf16*)take((size_t)H * H * 2);
  __bf16* w3p   = (__bf16*)take((size_t)NOUTP * H * 2);

  float* outRot   = (float*)d_out;                  // 8192*24*9
  float* outBetas = outRot + (size_t)Mrows * 216;   // 8192*10

  const int total = Mrows * H;
  const int nmask = H - 1;

  // ---- Pack weights to bf16 [Npad][Kp] ----
  pack_w_kernel<<<dim3((Kp0 + 255) / 256, H), 256, 0, stream>>>(W0, w0p, K0, H, Kp0);
  pack_w_kernel<<<dim3(H / 256, H), 256, 0, stream>>>(W1a, w1ap, H, H, H);
  pack_w_kernel<<<dim3(H / 256, H), 256, 0, stream>>>(W1b, w1bp, H, H, H);
  pack_w_kernel<<<dim3(H / 256, H), 256, 0, stream>>>(W2a, w2ap, H, H, H);
  pack_w_kernel<<<dim3(H / 256, H), 256, 0, stream>>>(W2b, w2bp, H, H, H);
  pack_w_kernel<<<dim3(H / 256, NOUTP), 256, 0, stream>>>(W3, w3p, H, NOUT, H);

  auto bnstats = [&](const float* yb) {
    bn_partial_kernel<<<dim3(H / 64, CHUNKS), 256, 0, stream>>>(yb, pbuf, Mrows, H, CHUNKS);
    bn_finalize_kernel<<<(H + 255) / 256, 256, 0, stream>>>(pbuf, stats, H, CHUNKS, Mrows);
  };

  const dim3 ggrid(H / BN, Mrows / BM);

  // ---- Layer 0: h = relu(bn(x @ W0 + b0)) ----
  gemm_kernel<false><<<ggrid, THREADS, 0, stream>>>(
      (const void*)x, w0p, b0, y, Mrows, H, K0, Kp0);
  bnstats(y);
  bn_apply_kernel<true, false><<<total / 256, 256, 0, stream>>>(
      y, stats, g0, be0, nullptr, h, hbf, total, nmask);

  // ---- Residual block 1 ----
  gemm_kernel<true><<<ggrid, THREADS, 0, stream>>>(
      (const void*)hbf, w1ap, b1a, y, Mrows, H, H, H);
  bnstats(y);
  bn_apply_kernel<true, false><<<total / 256, 256, 0, stream>>>(
      y, stats, g1a, be1a, nullptr, y, tbf, total, nmask);   // fp32 t unused -> alias y
  gemm_kernel<true><<<ggrid, THREADS, 0, stream>>>(
      (const void*)tbf, w1bp, b1b, y, Mrows, H, H, H);
  bnstats(y);
  bn_apply_kernel<true, true><<<total / 256, 256, 0, stream>>>(
      y, stats, g1b, be1b, h, h, hbf, total, nmask);

  // ---- Residual block 2 ----
  gemm_kernel<true><<<ggrid, THREADS, 0, stream>>>(
      (const void*)hbf, w2ap, b2a, y, Mrows, H, H, H);
  bnstats(y);
  bn_apply_kernel<true, false><<<total / 256, 256, 0, stream>>>(
      y, stats, g2a, be2a, nullptr, y, tbf, total, nmask);
  gemm_kernel<true><<<ggrid, THREADS, 0, stream>>>(
      (const void*)tbf, w2bp, b2b, y, Mrows, H, H, H);
  bnstats(y);
  bn_apply_kernel<true, true><<<total / 256, 256, 0, stream>>>(
      y, stats, g2b, be2b, h, h, hbf, total, nmask);

  // ---- Output head: yout = h @ W3 + b3 ----
  gemm_kernel<true><<<dim3((NOUT + BN - 1) / BN, Mrows / BM), THREADS, 0, stream>>>(
      (const void*)hbf, w3p, b3, yout, Mrows, NOUT, H, H);

  // ---- Betas slice + polar-orthogonalized rotations ----
  betas_kernel<<<(Mrows * 10 + 255) / 256, 256, 0, stream>>>(yout, outBetas, Mrows, NOUT);
  polar_kernel<<<(Mrows * 24 + 255) / 256, 256, 0, stream>>>(yout, outRot, Mrows * 24, NOUT);
}
